// SelfAttention_10428180595139
// MI455X (gfx1250) — matrix operand order repaired and checked by reading, not compiled, not run on hardware
//
#include <hip/hip_runtime.h>
#include <hip/hip_bf16.h>

// Self-attention forward for MI455X (gfx1250, wave32, WMMA).
// All GEMMs use v_wmma_f32_16x16x32_f16 (f16 operands, fp32 accumulate).
// Wave tile: 32x64 (2 A fragments x 4 B tiles, 8 accumulators) for 2x B reuse.

typedef __attribute__((ext_vector_type(16))) _Float16 v16h;
typedef __attribute__((ext_vector_type(8)))  _Float16 v8h;
typedef __attribute__((ext_vector_type(8)))  float    v8f;

#define T_DIM 4096
#define D_DIM 1024
#define H_DIM 8
#define HD_DIM 128

__device__ __forceinline__ v8f wmma_f16(v16h a, v16h b, v8f c) {
  return __builtin_amdgcn_wmma_f32_16x16x32_f16(
      /*neg_a=*/false, a, /*neg_b=*/false, b,
      /*c_mod=*/(short)0, c, /*reuse_a=*/false, /*reuse_b=*/false);
}

// A fragment: 16x32 f16 tile from row-major [.., ld] at (m0, k0).
// Lane L<16: M=L, K = {k0..k0+7, k0+16..k0+23}; lane L>=16: M=L-16, K shifted +8.
__device__ __forceinline__ v16h load_a16(const _Float16* __restrict__ base,
                                         int ld, int m0, int k0) {
  int lane = threadIdx.x & 31;
  int m  = m0 + (lane & 15);
  int kb = k0 + ((lane & 16) ? 8 : 0);
  const _Float16* p = base + (size_t)m * ld + kb;
  v8h lo = *(const v8h*)(p);        // 16B aligned
  v8h hi = *(const v8h*)(p + 16);
  v16h a;
#pragma unroll
  for (int e = 0; e < 8; ++e) { a[e] = lo[e]; a[e + 8] = hi[e]; }
  return a;
}

// A fragment from an fp32 row-major matrix, converted to f16 on load.
__device__ __forceinline__ v16h load_a32(const float* __restrict__ base,
                                         int ld, int m0, int k0) {
  int lane = threadIdx.x & 31;
  int m  = m0 + (lane & 15);
  int kb = k0 + ((lane & 16) ? 8 : 0);
  const float* p = base + (size_t)m * ld + kb;
  v16h a;
#pragma unroll
  for (int e = 0; e < 8; ++e) {
    a[e]     = (_Float16)p[e];
    a[e + 8] = (_Float16)p[e + 16];
  }
  return a;
}

// B fragment: 32x16 f16 tile where B(k,n) = src[n*ld + k] (k contiguous).
// Lane L<16: N=L, K = {k0..k0+7, k0+16..k0+23}; lane L>=16: N=L-16, K shifted +8.
__device__ __forceinline__ v16h load_b16(const _Float16* __restrict__ base,
                                         int ld, int n0, int k0) {
  int lane = threadIdx.x & 31;
  int n  = n0 + (lane & 15);
  int kb = k0 + ((lane & 16) ? 8 : 0);
  const _Float16* p = base + (size_t)n * ld + kb;
  v8h lo = *(const v8h*)(p);        // 16B aligned
  v8h hi = *(const v8h*)(p + 16);
  v16h b;
#pragma unroll
  for (int e = 0; e < 8; ++e) { b[e] = lo[e]; b[e + 8] = hi[e]; }
  return b;
}

// C/D layout: element v of accumulator = (M = v + 8*(lane>=16), N = lane&15).
__device__ __forceinline__ void store_c_f32(float* __restrict__ base, int ld,
                                            int m0, int n0, v8f c) {
  int lane = threadIdx.x & 31;
  int n  = n0 + (lane & 15);
  int mb = m0 + ((lane & 16) ? 8 : 0);
#pragma unroll
  for (int v = 0; v < 8; ++v) base[(size_t)(mb + v) * ld + n] = c[v];
}

__device__ __forceinline__ void store_c_f16(_Float16* __restrict__ base, int ld,
                                            int m0, int n0, v8f c) {
  int lane = threadIdx.x & 31;
  int n  = n0 + (lane & 15);
  int mb = m0 + ((lane & 16) ? 8 : 0);
#pragma unroll
  for (int v = 0; v < 8; ++v) base[(size_t)(mb + v) * ld + n] = (_Float16)c[v];
}

// Transposed f16 store: out[n][m] (used to build V^T once).
__device__ __forceinline__ void store_c_f16_T(_Float16* __restrict__ base, int ldT,
                                              int m0, int n0, v8f c) {
  int lane = threadIdx.x & 31;
  int n  = n0 + (lane & 15);
  int mb = m0 + ((lane & 16) ? 8 : 0);
#pragma unroll
  for (int v = 0; v < 8; ++v) base[(size_t)n * ldT + (mb + v)] = (_Float16)c[v];
}

__device__ __forceinline__ v8f zero8() {
  v8f z = {0.f, 0.f, 0.f, 0.f, 0.f, 0.f, 0.f, 0.f};
  return z;
}

// -------------------- kernels --------------------

__global__ __launch_bounds__(256) void cvt_f32_f16(const float* __restrict__ src,
                                                   _Float16* __restrict__ dst, int n) {
  int i = blockIdx.x * 256 + threadIdx.x;
  if (i < n) dst[i] = (_Float16)src[i];
}

// QKV projections. grid = (32, 24). z = proj*8 + h. Wave: 32x64 tile of [T,HD].
// Q,K stored row-major [H][T][HD]; V stored transposed [H][HD][T].
__global__ __launch_bounds__(256) void proj_kernel(const _Float16* __restrict__ xh,
                                                   const _Float16* __restrict__ wh,
                                                   _Float16* __restrict__ qh,
                                                   _Float16* __restrict__ kh,
                                                   _Float16* __restrict__ vth) {
  int z = blockIdx.y;
  int proj = z >> 3, h = z & 7;
  int wave = (blockIdx.x * 256 + threadIdx.x) >> 5;   // 0..255
  int m0 = (wave >> 1) * 32;
  int n0 = (wave & 1) * 64;
  const _Float16* w = wh + ((size_t)proj * H_DIM + h) * HD_DIM * D_DIM;  // [HD, D]
  v8f c0[4], c1[4];
#pragma unroll
  for (int j = 0; j < 4; ++j) { c0[j] = zero8(); c1[j] = zero8(); }
  for (int k0 = 0; k0 < D_DIM; k0 += 32) {
    v16h a0 = load_a16(xh, D_DIM, m0, k0);
    v16h a1 = load_a16(xh, D_DIM, m0 + 16, k0);
#pragma unroll
    for (int j = 0; j < 4; ++j) {
      v16h b = load_b16(w, D_DIM, n0 + j * 16, k0);  // B(k,n)=W[n][k]
      c0[j] = wmma_f16(a0, b, c0[j]);
      c1[j] = wmma_f16(a1, b, c1[j]);
    }
  }
  if (proj == 0) {
    _Float16* dst = qh + (size_t)h * T_DIM * HD_DIM;
#pragma unroll
    for (int j = 0; j < 4; ++j) {
      store_c_f16(dst, HD_DIM, m0, n0 + j * 16, c0[j]);
      store_c_f16(dst, HD_DIM, m0 + 16, n0 + j * 16, c1[j]);
    }
  } else if (proj == 1) {
    _Float16* dst = kh + (size_t)h * T_DIM * HD_DIM;
#pragma unroll
    for (int j = 0; j < 4; ++j) {
      store_c_f16(dst, HD_DIM, m0, n0 + j * 16, c0[j]);
      store_c_f16(dst, HD_DIM, m0 + 16, n0 + j * 16, c1[j]);
    }
  } else {
    _Float16* dst = vth + (size_t)h * HD_DIM * T_DIM;
#pragma unroll
    for (int j = 0; j < 4; ++j) {
      store_c_f16_T(dst, T_DIM, m0, n0 + j * 16, c0[j]);
      store_c_f16_T(dst, T_DIM, m0 + 16, n0 + j * 16, c1[j]);
    }
  }
}

// scores = Q @ K^T for one head -> fp32 [T,T]. 8192 waves -> 1024 blocks of 256.
__global__ __launch_bounds__(256) void scores_kernel(const _Float16* __restrict__ qh,
                                                     const _Float16* __restrict__ kh,
                                                     float* __restrict__ scores, int h) {
  const _Float16* Q = qh + (size_t)h * T_DIM * HD_DIM;
  const _Float16* K = kh + (size_t)h * T_DIM * HD_DIM;
  int wave = (blockIdx.x * 256 + threadIdx.x) >> 5;   // 0..8191
  int m0 = (wave >> 6) * 32;
  int n0 = (wave & 63) * 64;
  v8f c0[4], c1[4];
#pragma unroll
  for (int j = 0; j < 4; ++j) { c0[j] = zero8(); c1[j] = zero8(); }
#pragma unroll
  for (int k0 = 0; k0 < HD_DIM; k0 += 32) {
    v16h a0 = load_a16(Q, HD_DIM, m0, k0);
    v16h a1 = load_a16(Q, HD_DIM, m0 + 16, k0);
#pragma unroll
    for (int j = 0; j < 4; ++j) {
      v16h b = load_b16(K, HD_DIM, n0 + j * 16, k0);  // B(k,n)=K[n][k]
      c0[j] = wmma_f16(a0, b, c0[j]);
      c1[j] = wmma_f16(a1, b, c1[j]);
    }
  }
#pragma unroll
  for (int j = 0; j < 4; ++j) {
    store_c_f32(scores, T_DIM, m0, n0 + j * 16, c0[j]);
    store_c_f32(scores, T_DIM, m0 + 16, n0 + j * 16, c1[j]);
  }
}

// One block per row: add positional bias, softmax in registers, write probs in place.
// If lastHead != 0, also write the row to att_out (the reference returns att[-1]).
__global__ __launch_bounds__(256) void softmax_bias_kernel(float* __restrict__ scores,
                                                           float* __restrict__ att_out,
                                                           int lastHead) {
  const int r = blockIdx.x;
  const int t = threadIdx.x;
  float* row = scores + (size_t)r * T_DIM;
  __shared__ float red[8];
  float vals[16];
  const float NEG_L2F_OVER_D = -13.2877123795494f / 1024.0f;  // -log2(10000)/1024
  float mx = -3.4e38f;
#pragma unroll
  for (int i = 0; i < 16; ++i) {
    int s = t + i * 256;
    int ii = s >> 1;
    float ang = (float)r * exp2f((float)(2 * ii) * NEG_L2F_OVER_D);
    float bias = (s & 1) ? cosf(ang) : sinf(ang);
    float v = row[s] + bias;
    vals[i] = v;
    mx = fmaxf(mx, v);
  }
  for (int off = 16; off; off >>= 1) mx = fmaxf(mx, __shfl_xor(mx, off, 32));
  int w = t >> 5;
  if ((t & 31) == 0) red[w] = mx;
  __syncthreads();
  mx = red[0];
#pragma unroll
  for (int i = 1; i < 8; ++i) mx = fmaxf(mx, red[i]);
  __syncthreads();
  float sum = 0.f;
#pragma unroll
  for (int i = 0; i < 16; ++i) {
    vals[i] = __expf(vals[i] - mx);
    sum += vals[i];
  }
  for (int off = 16; off; off >>= 1) sum += __shfl_xor(sum, off, 32);
  if ((t & 31) == 0) red[w] = sum;
  __syncthreads();
  sum = 0.f;
#pragma unroll
  for (int i = 0; i < 8; ++i) sum += red[i];
  float inv = 1.0f / sum;
#pragma unroll
  for (int i = 0; i < 16; ++i) {
    int s = t + i * 256;
    float p = vals[i] * inv;
    row[s] = p;
    if (lastHead) att_out[(size_t)r * T_DIM + s] = p;
  }
}

// y[h] = att @ V[h]. A = fp32 att converted on load; B from V^T [HD][T].
// Writes into concatenated f16 y [T, D] at column h*HD. 256 waves -> 32 blocks.
__global__ __launch_bounds__(256) void av_kernel(const float* __restrict__ att,
                                                 const _Float16* __restrict__ vth,
                                                 _Float16* __restrict__ yh, int h) {
  const _Float16* Vt = vth + (size_t)h * HD_DIM * T_DIM;
  int wave = (blockIdx.x * 256 + threadIdx.x) >> 5;   // 0..255
  int m0  = (wave >> 1) * 32;
  int n0c = (wave & 1) * 64;
  v8f c0[4], c1[4];
#pragma unroll
  for (int j = 0; j < 4; ++j) { c0[j] = zero8(); c1[j] = zero8(); }
  for (int k0 = 0; k0 < T_DIM; k0 += 32) {
    v16h a0 = load_a32(att, T_DIM, m0, k0);
    v16h a1 = load_a32(att, T_DIM, m0 + 16, k0);
#pragma unroll
    for (int j = 0; j < 4; ++j) {
      v16h b = load_b16(Vt, T_DIM, n0c + j * 16, k0);  // B(k,n)=V[k][n]=Vt[n][k]
      c0[j] = wmma_f16(a0, b, c0[j]);
      c1[j] = wmma_f16(a1, b, c1[j]);
    }
  }
#pragma unroll
  for (int j = 0; j < 4; ++j) {
    store_c_f16(yh, D_DIM, m0, h * HD_DIM + n0c + j * 16, c0[j]);
    store_c_f16(yh, D_DIM, m0 + 16, h * HD_DIM + n0c + j * 16, c1[j]);
  }
}

// out = y @ Wout^T -> fp32 d_out [T, D]. 2048 waves -> 256 blocks.
__global__ __launch_bounds__(256) void out_kernel(const _Float16* __restrict__ yh,
                                                  const _Float16* __restrict__ wouth,
                                                  float* __restrict__ out) {
  int wave = (blockIdx.x * 256 + threadIdx.x) >> 5;   // 0..2047
  int m0 = (wave >> 4) * 32;
  int n0 = (wave & 15) * 64;
  v8f c0[4], c1[4];
#pragma unroll
  for (int j = 0; j < 4; ++j) { c0[j] = zero8(); c1[j] = zero8(); }
  for (int k0 = 0; k0 < D_DIM; k0 += 32) {
    v16h a0 = load_a16(yh, D_DIM, m0, k0);
    v16h a1 = load_a16(yh, D_DIM, m0 + 16, k0);
#pragma unroll
    for (int j = 0; j < 4; ++j) {
      v16h b = load_b16(wouth, D_DIM, n0 + j * 16, k0);  // B(k,n)=Wout[n][k]
      c0[j] = wmma_f16(a0, b, c0[j]);
      c1[j] = wmma_f16(a1, b, c1[j]);
    }
  }
#pragma unroll
  for (int j = 0; j < 4; ++j) {
    store_c_f32(out, D_DIM, m0, n0 + j * 16, c0[j]);
    store_c_f32(out, D_DIM, m0 + 16, n0 + j * 16, c1[j]);
  }
}

// -------------------- host launcher --------------------

extern "C" void kernel_launch(void* const* d_in, const int* in_sizes, int n_in,
                              void* d_out, int out_size, void* d_ws, size_t ws_size,
                              hipStream_t stream) {
  (void)in_sizes; (void)n_in; (void)out_size; (void)ws_size;
  const float* x    = (const float*)d_in[0];
  const float* Wq   = (const float*)d_in[1];
  const float* Wk   = (const float*)d_in[2];
  const float* Wv   = (const float*)d_in[3];
  const float* Wout = (const float*)d_in[4];

  float* out_y   = (float*)d_out;                       // [T, D]
  float* out_att = out_y + (size_t)T_DIM * D_DIM;       // [T, T] (last head)

  char* ws = (char*)d_ws;
  size_t off = 0;
  auto carve = [&](size_t bytes) -> void* {
    void* p = ws + off;
    off += (bytes + 255) & ~(size_t)255;
    return p;
  };
  _Float16* xh    = (_Float16*)carve((size_t)T_DIM * D_DIM * 2);
  _Float16* wh    = (_Float16*)carve((size_t)3 * H_DIM * HD_DIM * D_DIM * 2);
  _Float16* wouth = (_Float16*)carve((size_t)D_DIM * D_DIM * 2);
  _Float16* qh    = (_Float16*)carve((size_t)H_DIM * T_DIM * HD_DIM * 2);
  _Float16* kh    = (_Float16*)carve((size_t)H_DIM * T_DIM * HD_DIM * 2);
  _Float16* vth   = (_Float16*)carve((size_t)H_DIM * HD_DIM * T_DIM * 2);
  _Float16* yh    = (_Float16*)carve((size_t)T_DIM * D_DIM * 2);
  float*    sc    = (float*)carve((size_t)T_DIM * T_DIM * 4);

  const int WHD = H_DIM * HD_DIM * D_DIM;   // 1M elements per W{q,k,v}
  auto cvt = [&](const float* s, _Float16* d, int n) {
    cvt_f32_f16<<<(n + 255) / 256, 256, 0, stream>>>(s, d, n);
  };
  cvt(x, xh, T_DIM * D_DIM);
  cvt(Wq, wh, WHD);
  cvt(Wk, wh + (size_t)WHD, WHD);
  cvt(Wv, wh + (size_t)2 * WHD, WHD);
  cvt(Wout, wouth, D_DIM * D_DIM);

  dim3 gp(32, 24);
  proj_kernel<<<gp, 256, 0, stream>>>(xh, wh, qh, kh, vth);

  for (int h = 0; h < H_DIM; ++h) {
    scores_kernel<<<1024, 256, 0, stream>>>(qh, kh, sc, h);
    softmax_bias_kernel<<<T_DIM, 256, 0, stream>>>(sc, out_att, (h == H_DIM - 1) ? 1 : 0);
    av_kernel<<<32, 256, 0, stream>>>(sc, vth, yh, h);
  }

  out_kernel<<<256, 256, 0, stream>>>(yh, wouth, out_y);
}
